// HippocampalLSTM_13812614824016
// MI455X (gfx1250) — matrix operand order repaired and checked by reading, not compiled, not run on hardware
//
#include <hip/hip_runtime.h>

// ---------------- problem constants ----------------
#define BB 512
#define TT 1024
#define DD 14
#define HH 128

typedef __attribute__((ext_vector_type(16))) _Float16 v16h;
typedef __attribute__((ext_vector_type(8)))  _Float16 v8h;
typedef __attribute__((ext_vector_type(8)))  float    v8f;

// ---------------- LDS layout (units: _Float16) ----------------
// Padded row strides chosen so 16 lanes reading rows r..r+15 with ds_load_b128
// land on disjoint LDS bank groups (stride 272B = 68 dwords -> banks 4r..4r+3).
#define WROW 136                       // 128 cols + 8 pad  (272 B rows, 16B aligned)
#define XROW 24                        // 16 cols + 8 pad   (48 B rows, 16B aligned)
#define OFF_WH0  0
#define OFF_WX1  (OFF_WH0 + 512*WROW)
#define OFF_WX0  (OFF_WX1 + 512*WROW)  // padded [512][16]: cols 0..13 Wx0, col 14 = bx0, 15 = 0
#define OFF_H0   (OFF_WX0 + 512*XROW)
#define OFF_H1   (OFF_H0  + 16*WROW)
#define OFF_XT   (OFF_H1  + 16*WROW)   // x tile [16][16]: cols 0..13 x_t, col 14 = 1.0 (bias lane)
#define OFF_WO   (OFF_XT  + 16*XROW)   // 128 floats (256 halves)
#define SMEM_HALVES (OFF_WO + 256)
#define SMEM_BYTES  (SMEM_HALVES * 2)  // 313088 B < 320 KB WGP LDS

// ---------------- fast math helpers ----------------
static __device__ __forceinline__ float fexp2(float x) {
#if __has_builtin(__builtin_amdgcn_exp2f)
  return __builtin_amdgcn_exp2f(x);
#else
  return __builtin_exp2f(x);
#endif
}
static __device__ __forceinline__ float frcp(float x) {
#if __has_builtin(__builtin_amdgcn_rcpf)
  return __builtin_amdgcn_rcpf(x);
#else
  return 1.0f / x;
#endif
}
static __device__ __forceinline__ float sigf(float x) {
  // 1/(1+e^-x) via v_exp_f32 (exp2) + v_rcp_f32
  return frcp(1.0f + fexp2(-1.4426950408889634f * x));
}
static __device__ __forceinline__ float tanh_f(float x) {
  return 2.0f * sigf(2.0f * x) - 1.0f;
}

static __device__ __forceinline__ v8f wmma16(v16h a, v16h b, v8f c) {
  // (neg_a, A, neg_b, B, c_mod, C, reuse_a, reuse_b)
  return __builtin_amdgcn_wmma_f32_16x16x32_f16(false, a, false, b, (short)0, c,
                                                false, false);
}

// A-fragment (16x32 f16) from a padded-row f16 tile:
// lane L: M = L&15 ; halves 0..15 = K kb..kb+15, kb = (L>=16)*16
static __device__ __forceinline__ v16h loadA(const _Float16* tile, int row_off,
                                             int ks, int kb) {
  union { v16h v; v8h p[2]; } u;
  const _Float16* p = tile + row_off + ks * 32 + kb;
  u.p[0] = *(const v8h*)(p);
  u.p[1] = *(const v8h*)(p + 8);
  return u.v;
}
// B-fragment (32x16 f16) from padded row-major weights W[4H][128]:
// lane L: N = L&15 (weight row), halves 0..15 = K kseg..kseg+15, kseg = (L>=16)*16
static __device__ __forceinline__ v16h loadB(const _Float16* w, int row, int ks,
                                             int kseg) {
  union { v16h v; v8h p[2]; } u;
  const _Float16* p = w + row * WROW + ks * 32 + kseg;
  u.p[0] = *(const v8h*)(p);
  u.p[1] = *(const v8h*)(p + 8);
  return u.v;
}

__launch_bounds__(256, 1)
__global__ void lstm2_kernel(const float* __restrict__ xg,
                             const float* __restrict__ gWx0,
                             const float* __restrict__ gbx0,
                             const float* __restrict__ gWh0,
                             const float* __restrict__ gWx1,
                             const float* __restrict__ gbx1,
                             const float* __restrict__ gWh1,
                             const float* __restrict__ gWo,
                             const float* __restrict__ gbo,
                             float* __restrict__ outp,
                             float* __restrict__ hidp) {
  extern __shared__ _Float16 smem[];
  _Float16* Lwh0 = smem + OFF_WH0;
  _Float16* Lwx1 = smem + OFF_WX1;
  _Float16* Lwx0 = smem + OFF_WX0;
  _Float16* Lh0  = smem + OFF_H0;
  _Float16* Lh1  = smem + OFF_H1;
  _Float16* Lxt  = smem + OFF_XT;
  float*    Lwo  = (float*)(smem + OFF_WO);

  const int tid  = threadIdx.x;
  const int wv   = tid >> 5;        // wave 0..7 -> owns hidden units [16w,16w+16)
  const int lane = tid & 31;
  const int ln   = lane & 15;
  const int lh   = lane >> 4;       // half-wave id
  const int w16  = wv * 16;
  const int kb   = lh * 16;         // A fragment K base
  const int ksg  = lh * 16;         // B fragment K base
  const int moff = lh * 8;          // C/D fragment M base
  const int b0   = blockIdx.x * 16; // batch tile

  // ---------------- prologue: stage f16 weights on-chip ----------------
  for (int i = tid; i < 512 * HH; i += 256) {
    int r = i >> 7, c = i & 127;
    Lwh0[r * WROW + c] = (_Float16)gWh0[i];
    Lwx1[r * WROW + c] = (_Float16)gWx1[i];
  }
  for (int i = tid; i < 512 * XROW; i += 256) Lwx0[i] = (_Float16)0.0f;
  for (int i = tid; i < 16 * WROW; i += 256) {
    Lh0[i] = (_Float16)0.0f;
    Lh1[i] = (_Float16)0.0f;
  }
  for (int i = tid; i < 16 * XROW; i += 256) Lxt[i] = (_Float16)0.0f;
  for (int i = tid; i < HH; i += 256) Lwo[i] = gWo[i];
  __syncthreads();
  for (int i = tid; i < 512 * DD; i += 256) {
    int r = i / DD, c = i - r * DD;
    Lwx0[r * XROW + c] = (_Float16)gWx0[i];
  }
  for (int r = tid; r < 512; r += 256) Lwx0[r * XROW + DD] = (_Float16)gbx0[r];
  for (int m = tid; m < 16; m += 256) Lxt[m * XROW + DD] = (_Float16)1.0f; // bias lane

  // Wh1: VGPR-resident B fragments (16 x v16h = 128 VGPRs per wave) — zero
  // in-loop memory traffic for the layer-1 recurrent weights.
  v16h wh1B[16];
#pragma unroll
  for (int q = 0; q < 4; ++q) {
#pragma unroll
    for (int ks = 0; ks < 4; ++ks) {
      const float* src = gWh1 + (size_t)(q * HH + w16 + ln) * HH + ks * 32 + ksg;
      v16h v;
#pragma unroll
      for (int j = 0; j < 16; ++j) v[j] = (_Float16)src[j];
      wh1B[q * 4 + ks] = v;
    }
  }
  float b1[4];
#pragma unroll
  for (int q = 0; q < 4; ++q) b1[q] = gbx1[q * HH + w16 + ln];
  const float bo0 = gbo[0];

  // cell states live in registers in the WMMA C/D layout
  float c0[8], c1[8];
#pragma unroll
  for (int r = 0; r < 8; ++r) { c0[r] = 0.0f; c1[r] = 0.0f; }

  __syncthreads();

  // ---------------- sequential time loop ----------------
  for (int t = 0; t < TT; ++t) {
    // stage x_t tile (cols 0..13; col 14 stays 1.0 -> folds bx0 into the x-WMMA)
    if (tid < 16 * DD) {
      int m = tid / DD, k = tid - m * DD;
      Lxt[m * XROW + k] = (_Float16)xg[((size_t)(b0 + m) * TT + t) * DD + k];
    }
    __syncthreads();  // (A) x staged, previous-step h writes visible

    // ----- layer 0: gates = [x_t,1] @ [Wx0,bx0]^T + h0 @ Wh0^T -----
    v16h aX;
    {
      union { v16h v; v8h p[2]; } u;
      v8h z = {};
      if (lh == 0) {  // K 0..15 (D=14 + bias + pad)
        u.p[0] = *(const v8h*)(Lxt + ln * XROW);
        u.p[1] = *(const v8h*)(Lxt + ln * XROW + 8);
      } else {        // K 16..31 are zero
        u.p[0] = z; u.p[1] = z;
      }
      aX = u.v;
    }
    v16h aH[4];
#pragma unroll
    for (int ks = 0; ks < 4; ++ks) aH[ks] = loadA(Lh0, ln * WROW, ks, kb);

    v8f acc[4];
#pragma unroll
    for (int q = 0; q < 4; ++q) {
      union { v16h v; v8h p[2]; } ub;
      if (lh == 0) {  // lanes 16..31 hold K 16..31 of B = zero
        const _Float16* p = Lwx0 + (q * HH + w16 + ln) * XROW;
        ub.p[0] = *(const v8h*)(p);
        ub.p[1] = *(const v8h*)(p + 8);
      } else {
        v8h z = {};
        ub.p[0] = z; ub.p[1] = z;
      }
      v8f c = {};
      c = wmma16(aX, ub.v, c);
#pragma unroll
      for (int ks = 0; ks < 4; ++ks)
        c = wmma16(aH[ks], loadB(Lwh0, q * HH + w16 + ln, ks, ksg), c);
      acc[q] = c;
    }
    __syncthreads();  // (B) everyone done reading old Lh0 / Lxt

    float h0n[8];
#pragma unroll
    for (int r = 0; r < 8; ++r) {
      float iv = sigf(acc[0][r]);
      float fv = sigf(acc[1][r]);
      float gv = tanh_f(acc[2][r]);
      float ov = sigf(acc[3][r]);
      float cn = fv * c0[r] + iv * gv;
      c0[r] = cn;
      h0n[r] = ov * tanh_f(cn);
    }
#pragma unroll
    for (int r = 0; r < 8; ++r) {
      int m = r + moff;
      Lh0[m * WROW + w16 + ln] = (_Float16)h0n[r];
      hidp[((size_t)(b0 + m) * TT + t) * (2 * HH) + (w16 + ln)] = h0n[r];
    }
    __syncthreads();  // (C) new h0 visible

    // ----- layer 1: gates = h0_new @ Wx1^T + bx1 + h1 @ Wh1^T -----
    v16h aH0n[4], aH1[4];
#pragma unroll
    for (int ks = 0; ks < 4; ++ks) {
      aH0n[ks] = loadA(Lh0, ln * WROW, ks, kb);
      aH1[ks]  = loadA(Lh1, ln * WROW, ks, kb);
    }
#pragma unroll
    for (int q = 0; q < 4; ++q) {
      v8f c = {};
#pragma unroll
      for (int ks = 0; ks < 4; ++ks) {
        c = wmma16(aH0n[ks], loadB(Lwx1, q * HH + w16 + ln, ks, ksg), c);
        c = wmma16(aH1[ks], wh1B[q * 4 + ks], c);
      }
      acc[q] = c;
    }
    __syncthreads();  // (D) everyone done reading old Lh1

    float h1n[8];
#pragma unroll
    for (int r = 0; r < 8; ++r) {
      float iv = sigf(acc[0][r] + b1[0]);
      float fv = sigf(acc[1][r] + b1[1]);
      float gv = tanh_f(acc[2][r] + b1[2]);
      float ov = sigf(acc[3][r] + b1[3]);
      float cn = fv * c1[r] + iv * gv;
      c1[r] = cn;
      h1n[r] = ov * tanh_f(cn);
    }
#pragma unroll
    for (int r = 0; r < 8; ++r) {
      int m = r + moff;
      Lh1[m * WROW + w16 + ln] = (_Float16)h1n[r];
      hidp[((size_t)(b0 + m) * TT + t) * (2 * HH) + HH + (w16 + ln)] = h1n[r];
    }
    __syncthreads();  // (E) new h1 visible

    // ----- output head: out[b,t] = sigmoid(h1 . Wo + bo), wave 0 only -----
    if (wv == 0) {
      const _Float16* hr = Lh1 + ln * WROW + lh * 64;
      const float* wo = Lwo + lh * 64;
      float s = 0.0f;
#pragma unroll 8
      for (int k = 0; k < 64; ++k) s += (float)hr[k] * wo[k];
      s += __shfl_xor(s, 16, 32);  // combine the two K halves
      if (lh == 0) outp[(size_t)(b0 + ln) * TT + t] = sigf(s + bo0);
    }
  }
}

extern "C" void kernel_launch(void* const* d_in, const int* in_sizes, int n_in,
                              void* d_out, int out_size, void* d_ws, size_t ws_size,
                              hipStream_t stream) {
  (void)in_sizes; (void)n_in; (void)out_size; (void)d_ws; (void)ws_size;
  const float* xg  = (const float*)d_in[0];
  const float* Wx0 = (const float*)d_in[1];
  const float* bx0 = (const float*)d_in[2];
  const float* Wh0 = (const float*)d_in[3];
  const float* Wx1 = (const float*)d_in[4];
  const float* bx1 = (const float*)d_in[5];
  const float* Wh1 = (const float*)d_in[6];
  const float* Wo  = (const float*)d_in[7];
  const float* bo  = (const float*)d_in[8];

  float* outp = (float*)d_out;                    // [B,T,1]
  float* hidp = outp + (size_t)BB * TT;           // [B,T,2H]

  (void)hipFuncSetAttribute((const void*)lstm2_kernel,
                            hipFuncAttributeMaxDynamicSharedMemorySize,
                            (int)SMEM_BYTES);
  lstm2_kernel<<<dim3(BB / 16), dim3(256), SMEM_BYTES, stream>>>(
      xg, Wx0, bx0, Wh0, Wx1, bx1, Wh1, Wo, bo, outp, hidp);
}
